// ConformerBlock_15315853377879
// MI455X (gfx1250) — matrix operand order repaired and checked by reading, not compiled, not run on hardware
//
#include <hip/hip_runtime.h>
#include <math.h>

typedef __attribute__((ext_vector_type(16))) _Float16 v16h;
typedef __attribute__((ext_vector_type(8)))  float    v8f;

union Frag16 {
  v16h v;
  _Float16 h[16];
  uint4 q[2];
};

#define LN_EPS 1e-5f

// --------------------------------------------------------------------------
// CDNA5 async global->LDS copy (16B per lane, ASYNCcnt-tracked) + waits.
// --------------------------------------------------------------------------
__device__ __forceinline__ void async_b128(unsigned lds_off, const _Float16* g) {
  asm volatile("global_load_async_to_lds_b128 %0, %1, off"
               :: "v"(lds_off), "v"((unsigned long long)(size_t)g)
               : "memory");
}
__device__ __forceinline__ void wait_async(int pending) {
  if (pending) asm volatile("s_wait_asynccnt 0x2" ::: "memory");
  else         asm volatile("s_wait_asynccnt 0x0" ::: "memory");
}

// --------------------------------------------------------------------------
// LayerNorm over last dim (512). One block (128 threads) per row.
// Writes f16 (for WMMA consumers) and optionally f32 (for chained LN).
// --------------------------------------------------------------------------
__global__ __launch_bounds__(128)
void ln_kernel(const float* __restrict__ in,
               const float* __restrict__ g,
               const float* __restrict__ b,
               _Float16* __restrict__ out16,
               float* __restrict__ out32) {
  __shared__ float rs[128];
  __shared__ float rq[128];
  const int row = blockIdx.x;
  const int t = threadIdx.x;
  const float* x = in + (size_t)row * 512;
  float v0[4];
  float s = 0.f, sq = 0.f;
#pragma unroll
  for (int i = 0; i < 4; ++i) {
    float f = x[i * 128 + t];
    v0[i] = f;
    s += f;
    sq += f * f;
  }
  rs[t] = s; rq[t] = sq;
  __syncthreads();
  for (int off = 64; off > 0; off >>= 1) {
    if (t < off) { rs[t] += rs[t + off]; rq[t] += rq[t + off]; }
    __syncthreads();
  }
  const float mean = rs[0] * (1.f / 512.f);
  const float var  = rq[0] * (1.f / 512.f) - mean * mean;
  const float rstd = rsqrtf(var + LN_EPS);
#pragma unroll
  for (int i = 0; i < 4; ++i) {
    const int c = i * 128 + t;
    const float v = (v0[i] - mean) * rstd * g[c] + b[c];
    out16[(size_t)row * 512 + c] = (_Float16)v;
    if (out32) out32[(size_t)row * 512 + c] = v;
  }
}

// --------------------------------------------------------------------------
// Weight cast + transpose: W[K][N] f32 -> Wt[N][K] f16 (once per launch).
// --------------------------------------------------------------------------
__global__ void wcast_kernel(const float* __restrict__ W,
                             _Float16* __restrict__ Wt, int K, int N) {
  const int i = blockIdx.x * blockDim.x + threadIdx.x;   // over K*N
  const int k = i / N;
  const int n = i - k * N;
  Wt[(size_t)n * K + k] = (_Float16)W[i];
}

// --------------------------------------------------------------------------
// WMMA GEMM:  C[M,N] = epilogue( A[M,K]f16 @ Wt[N,K]f16^T + bias[N] )
// Block = 256 threads (8 waves), tile 64x64, BK=32, double-buffered LDS
// staged entirely by global_load_async_to_lds_b128 (no VGPR staging, no cvt).
// Wave (wm 0..3, wn 0..1) -> rows wm*16..+16, cols wn*32..+32 (2 accums).
// epilogue: optional ReLU, optional resid + rscale*val, f16 or f32 output.
// --------------------------------------------------------------------------
__global__ __launch_bounds__(256)
void gemm_wmma_kernel(const _Float16* __restrict__ A,
                      const _Float16* __restrict__ Wt,
                      const float* __restrict__ bias,
                      const float* __restrict__ resid, float rscale, int has_resid,
                      float* __restrict__ C32, _Float16* __restrict__ C16, int out16,
                      int N, int K, int do_relu) {
  __shared__ _Float16 lA[2][64][32];
  __shared__ _Float16 lB[2][64][32];   // Wt tile, [n][k]

  const int t = threadIdx.x;
  const int lane = t & 31;
  const int wave = t >> 5;
  const int wm = wave >> 1;
  const int wn = wave & 1;
  const int row0 = blockIdx.y * 64;
  const int col0 = blockIdx.x * 64;
  const int khalf = lane >> 4;
  const int l15 = lane & 15;

  // each thread DMAs one 16B chunk of A and one of B per K-step
  const int cr = t >> 2;          // 0..63 : tile row (A) / tile n (B)
  const int cc = (t & 3) * 8;     // 0,8,16,24 halves (16B chunks)

  v8f acc0 = {}; v8f acc1 = {};
  const int nsteps = K >> 5;

  // prologue: DMA step 0 into buffer 0
  async_b128((unsigned)(size_t)&lA[0][cr][cc], A  + (size_t)(row0 + cr) * K + cc);
  async_b128((unsigned)(size_t)&lB[0][cr][cc], Wt + (size_t)(col0 + cr) * K + cc);

  for (int s = 0; s < nsteps; ++s) {
    const int buf = s & 1;
    if (s + 1 < nsteps) {
      const int k1 = (s + 1) << 5;
      async_b128((unsigned)(size_t)&lA[buf ^ 1][cr][cc], A  + (size_t)(row0 + cr) * K + k1 + cc);
      async_b128((unsigned)(size_t)&lB[buf ^ 1][cr][cc], Wt + (size_t)(col0 + cr) * K + k1 + cc);
      wait_async(1);              // wait current step's 2 DMAs; next 2 in flight
    } else {
      wait_async(0);
    }
    __syncthreads();

    Frag16 af, bf0, bf1;
    {
      const int r = wm * 16 + l15;
      af.q[0] = *(const uint4*)&lA[buf][r][khalf * 16];       // K = khalf*16+0..7
      af.q[1] = *(const uint4*)&lA[buf][r][khalf * 16 + 8];   // K = khalf*16+8..15
    }
    {
      const int n0 = wn * 32 + l15;
      bf0.q[0] = ((const uint4*)&lB[buf][n0][khalf * 16])[0]; // K = khalf*16+0..15
      bf0.q[1] = ((const uint4*)&lB[buf][n0][khalf * 16])[1];
      bf1.q[0] = ((const uint4*)&lB[buf][n0 + 16][khalf * 16])[0];
      bf1.q[1] = ((const uint4*)&lB[buf][n0 + 16][khalf * 16])[1];
    }
    acc0 = __builtin_amdgcn_wmma_f32_16x16x32_f16(false, af.v, false, bf0.v,
                                                  (short)0, acc0, false, false);
    acc1 = __builtin_amdgcn_wmma_f32_16x16x32_f16(false, af.v, false, bf1.v,
                                                  (short)0, acc1, false, false);
    __syncthreads();
  }

#pragma unroll
  for (int j = 0; j < 8; ++j) {
    const int r = row0 + wm * 16 + khalf * 8 + j;
#pragma unroll
    for (int tn = 0; tn < 2; ++tn) {
      const int c = col0 + wn * 32 + tn * 16 + l15;
      float v = (tn == 0 ? acc0[j] : acc1[j]) + bias[c];
      if (do_relu) v = fmaxf(v, 0.f);
      if (has_resid) v = resid[(size_t)r * N + c] + rscale * v;
      if (out16) C16[(size_t)r * N + c] = (_Float16)v;
      else       C32[(size_t)r * N + c] = v;
    }
  }
}

// --------------------------------------------------------------------------
// Flash attention on f16 Q/K/V (B=8 S=1024 H=8 DK=64). One wave per block,
// 16 query rows, online softmax over 32-key tiles. 8 WMMAs per key block.
// Fragments are raw uint4 copies (no conversions). 1/sqrt(DK) applied to
// f32 scores. Writes f16 O for the output projection GEMM.
// --------------------------------------------------------------------------
__global__ __launch_bounds__(32)
void flash_attn_kernel(const _Float16* __restrict__ Q,
                       const _Float16* __restrict__ Km,
                       const _Float16* __restrict__ V,
                       _Float16* __restrict__ O) {
  __shared__ _Float16 Plds[16][32];
  const int lane = threadIdx.x;
  const int l15 = lane & 15;
  const int khalf = lane >> 4;
  const int q0 = blockIdx.x * 16;
  const int bh = (int)blockIdx.y;
  const int b = bh >> 3;               // H = 8
  const int h = bh & 7;
  const size_t base = ((size_t)b * 1024) * 512 + (size_t)h * 64;

  Frag16 qf[2];
  {
    const _Float16* qrow = Q + base + (size_t)(q0 + l15) * 512;
#pragma unroll
    for (int ks = 0; ks < 2; ++ks) {
      qf[ks].q[0] = *(const uint4*)(qrow + ks * 32 + khalf * 16);
      qf[ks].q[1] = *(const uint4*)(qrow + ks * 32 + khalf * 16 + 8);
    }
  }

  float mrow[8], lrow[8];
#pragma unroll
  for (int j = 0; j < 8; ++j) { mrow[j] = -1e30f; lrow[j] = 0.f; }
  v8f o0 = {}, o1 = {}, o2 = {}, o3 = {};

  for (int kb = 0; kb < 1024; kb += 32) {
    v8f sc0 = {}, sc1 = {};
#pragma unroll
    for (int ks = 0; ks < 2; ++ks) {
      Frag16 kf0, kf1;
      const _Float16* kr0 = Km + base + (size_t)(kb + l15) * 512 + ks * 32 + khalf * 16;
      const _Float16* kr1 = Km + base + (size_t)(kb + 16 + l15) * 512 + ks * 32 + khalf * 16;
      kf0.q[0] = ((const uint4*)kr0)[0];
      kf0.q[1] = ((const uint4*)kr0)[1];
      kf1.q[0] = ((const uint4*)kr1)[0];
      kf1.q[1] = ((const uint4*)kr1)[1];
      sc0 = __builtin_amdgcn_wmma_f32_16x16x32_f16(false, qf[ks].v, false, kf0.v,
                                                   (short)0, sc0, false, false);
      sc1 = __builtin_amdgcn_wmma_f32_16x16x32_f16(false, qf[ks].v, false, kf1.v,
                                                   (short)0, sc1, false, false);
    }
    float pr0[8], pr1[8];
#pragma unroll
    for (int j = 0; j < 8; ++j) {
      const float s0 = sc0[j] * 0.125f;    // 1/sqrt(64)
      const float s1 = sc1[j] * 0.125f;
      float mx = fmaxf(s0, s1);
#pragma unroll
      for (int m = 1; m < 16; m <<= 1) mx = fmaxf(mx, __shfl_xor(mx, m, 32));
      const float mnew = fmaxf(mrow[j], mx);
      const float corr = __expf(mrow[j] - mnew);
      mrow[j] = mnew;
      const float p0 = __expf(s0 - mnew);
      const float p1 = __expf(s1 - mnew);
      float ps = p0 + p1;
#pragma unroll
      for (int m = 1; m < 16; m <<= 1) ps += __shfl_xor(ps, m, 32);
      lrow[j] = lrow[j] * corr + ps;
      o0[j] *= corr; o1[j] *= corr; o2[j] *= corr; o3[j] *= corr;
      pr0[j] = p0; pr1[j] = p1;
    }
#pragma unroll
    for (int j = 0; j < 8; ++j) {
      Plds[khalf * 8 + j][l15] = (_Float16)pr0[j];
      Plds[khalf * 8 + j][16 + l15] = (_Float16)pr1[j];
    }
    __syncthreads();
    Frag16 pf;
    pf.q[0] = *(const uint4*)&Plds[l15][khalf * 16];
    pf.q[1] = *(const uint4*)&Plds[l15][khalf * 16 + 8];
    __syncthreads();
#pragma unroll
    for (int tv = 0; tv < 4; ++tv) {
      Frag16 vf;
      const _Float16* vp = V + base + (size_t)(kb + khalf * 16) * 512 + tv * 16 + l15;
#pragma unroll
      for (int i = 0; i < 16; ++i)
        vf.h[i] = vp[(size_t)i * 512];
      if      (tv == 0) o0 = __builtin_amdgcn_wmma_f32_16x16x32_f16(false, pf.v, false, vf.v, (short)0, o0, false, false);
      else if (tv == 1) o1 = __builtin_amdgcn_wmma_f32_16x16x32_f16(false, pf.v, false, vf.v, (short)0, o1, false, false);
      else if (tv == 2) o2 = __builtin_amdgcn_wmma_f32_16x16x32_f16(false, pf.v, false, vf.v, (short)0, o2, false, false);
      else              o3 = __builtin_amdgcn_wmma_f32_16x16x32_f16(false, pf.v, false, vf.v, (short)0, o3, false, false);
    }
  }

#pragma unroll
  for (int j = 0; j < 8; ++j) {
    const float inv = 1.f / lrow[j];
    const size_t orow = base + (size_t)(q0 + khalf * 8 + j) * 512;
    O[orow + l15]      = (_Float16)(o0[j] * inv);
    O[orow + 16 + l15] = (_Float16)(o1[j] * inv);
    O[orow + 32 + l15] = (_Float16)(o2[j] * inv);
    O[orow + 48 + l15] = (_Float16)(o3[j] * inv);
  }
}

// --------------------------------------------------------------------------
// GLU: u[r,c] = hg[r,c] * sigmoid(hg[r, 512+c]),  hg is [8192,1024] f32
// --------------------------------------------------------------------------
__global__ void glu_kernel(const float* __restrict__ hg, float* __restrict__ u) {
  const int i = blockIdx.x * blockDim.x + threadIdx.x;
  const int row = i >> 9;
  const int c = i & 511;
  const float a = hg[(size_t)row * 1024 + c];
  const float gt = hg[(size_t)row * 1024 + 512 + c];
  u[i] = a * (1.f / (1.f + __expf(-gt)));
}

// --------------------------------------------------------------------------
// Depthwise conv (K=31, pad 15) + bias + BN (inference) + SiLU -> f16.
// --------------------------------------------------------------------------
__global__ void dwconv_bn_silu_kernel(const float* __restrict__ u,
                                      const float* __restrict__ w,   // [512][31]
                                      const float* __restrict__ wb,
                                      const float* __restrict__ bng,
                                      const float* __restrict__ bnb,
                                      const float* __restrict__ bnm,
                                      const float* __restrict__ bnv,
                                      _Float16* __restrict__ out) {
  const int i = blockIdx.x * blockDim.x + threadIdx.x;
  const int d = i & 511;
  const int s = (i >> 9) & 1023;
  const int b = i >> 19;
  float acc = wb[d];
  const float* wd = w + d * 31;
  const size_t rowbase = ((size_t)b * 1024) * 512 + d;
#pragma unroll
  for (int t = 0; t < 31; ++t) {
    const int ss = s + t - 15;
    if (ss >= 0 && ss < 1024)
      acc += u[rowbase + (size_t)ss * 512] * wd[t];
  }
  const float scale = bng[d] * rsqrtf(bnv[d] + LN_EPS);
  const float v = (acc - bnm[d]) * scale + bnb[d];
  out[i] = (_Float16)(v * (1.f / (1.f + __expf(-v))));
}

// --------------------------------------------------------------------------
// Launch sequence
// --------------------------------------------------------------------------
extern "C" void kernel_launch(void* const* d_in, const int* in_sizes, int n_in,
                              void* d_out, int out_size, void* d_ws, size_t ws_size,
                              hipStream_t stream) {
  const float* x     = (const float*)d_in[0];
  const float* n1g   = (const float*)d_in[1];
  const float* n1b   = (const float*)d_in[2];
  const float* n2g   = (const float*)d_in[3];
  const float* n2b   = (const float*)d_in[4];
  const float* n3g   = (const float*)d_in[5];
  const float* n3b   = (const float*)d_in[6];
  const float* n4g   = (const float*)d_in[7];
  const float* n4b   = (const float*)d_in[8];
  const float* clng  = (const float*)d_in[9];
  const float* clnb  = (const float*)d_in[10];
  const float* f1w1  = (const float*)d_in[11];
  const float* f1b1  = (const float*)d_in[12];
  const float* f1w2  = (const float*)d_in[13];
  const float* f1b2  = (const float*)d_in[14];
  const float* f2w1  = (const float*)d_in[15];
  const float* f2b1  = (const float*)d_in[16];
  const float* f2w2  = (const float*)d_in[17];
  const float* f2b2  = (const float*)d_in[18];
  const float* wq    = (const float*)d_in[19];
  const float* bq    = (const float*)d_in[20];
  const float* wk    = (const float*)d_in[21];
  const float* bk    = (const float*)d_in[22];
  const float* wv    = (const float*)d_in[23];
  const float* bv    = (const float*)d_in[24];
  const float* wo    = (const float*)d_in[25];
  const float* bo    = (const float*)d_in[26];
  const float* pw1w  = (const float*)d_in[27];
  const float* pw1b  = (const float*)d_in[28];
  const float* dww   = (const float*)d_in[29];
  const float* dwb   = (const float*)d_in[30];
  const float* bng   = (const float*)d_in[31];
  const float* bnbp  = (const float*)d_in[32];
  const float* bnm   = (const float*)d_in[33];
  const float* bnv   = (const float*)d_in[34];
  const float* pw2w  = (const float*)d_in[35];
  const float* pw2b  = (const float*)d_in[36];

  const int M = 8192;                                 // B*S
  char* wsb = (char*)d_ws;
  size_t off = 0;
  auto alloc = [&](size_t bytes) -> void* {
    void* p = wsb + off;
    off += (bytes + 255) & ~(size_t)255;
    return p;
  };
  const size_t R16 = (size_t)M * 512 * sizeof(_Float16);  // 8 MB
  const size_t R32 = (size_t)M * 512 * sizeof(float);     // 16 MB

  _Float16* h16 = (_Float16*)alloc(R16);
  float*    h32 = (float*)   alloc(R32);
  _Float16* t16 = (_Float16*)alloc(4 * R16);              // [8192,2048] f16
  _Float16* q16 = (_Float16*)alloc(R16);
  _Float16* k16 = (_Float16*)alloc(R16);
  _Float16* v16 = (_Float16*)alloc(R16);
  _Float16* o16 = (_Float16*)alloc(R16);
  float*    xa  = (float*)   alloc(R32);
  float*    hg  = (float*)   alloc(2 * R32);              // [8192,1024] f32
  float*    ub  = (float*)   alloc(R32);
  _Float16* c16 = (_Float16*)alloc(R16);
  // transposed f16 weights
  _Float16* f1w1t = (_Float16*)alloc((size_t)512 * 2048 * 2);
  _Float16* f1w2t = (_Float16*)alloc((size_t)2048 * 512 * 2);
  _Float16* f2w1t = (_Float16*)alloc((size_t)512 * 2048 * 2);
  _Float16* f2w2t = (_Float16*)alloc((size_t)2048 * 512 * 2);
  _Float16* wqt   = (_Float16*)alloc((size_t)512 * 512 * 2);
  _Float16* wkt   = (_Float16*)alloc((size_t)512 * 512 * 2);
  _Float16* wvt   = (_Float16*)alloc((size_t)512 * 512 * 2);
  _Float16* wot   = (_Float16*)alloc((size_t)512 * 512 * 2);
  _Float16* pw1t  = (_Float16*)alloc((size_t)512 * 1024 * 2);
  _Float16* pw2t  = (_Float16*)alloc((size_t)512 * 512 * 2);
  float* outp = (float*)d_out;

  auto wcast = [&](const float* W, _Float16* Wt, int K, int N) {
    wcast_kernel<<<(K * N) / 256, 256, 0, stream>>>(W, Wt, K, N);
  };
  wcast(f1w1, f1w1t, 512, 2048);
  wcast(f1w2, f1w2t, 2048, 512);
  wcast(f2w1, f2w1t, 512, 2048);
  wcast(f2w2, f2w2t, 2048, 512);
  wcast(wq, wqt, 512, 512);
  wcast(wk, wkt, 512, 512);
  wcast(wv, wvt, 512, 512);
  wcast(wo, wot, 512, 512);
  wcast(pw1w, pw1t, 512, 1024);
  wcast(pw2w, pw2t, 512, 512);

  auto gemm = [&](const _Float16* A, const _Float16* Wt, int N, int K,
                  const float* bias, const float* resid, float rscale,
                  float* C32, _Float16* C16, int relu) {
    dim3 grid(N / 64, M / 64);
    gemm_wmma_kernel<<<grid, 256, 0, stream>>>(A, Wt, bias,
                                               resid, rscale, resid != nullptr,
                                               C32, C16, C16 != nullptr,
                                               N, K, relu);
  };

  const int ew = 4 * 1024 * 1024;   // 8192*512 elementwise threads

  // ---- half-step FF1 ----
  ln_kernel<<<M, 128, 0, stream>>>(x, n1g, n1b, h16, nullptr);
  gemm(h16, f1w1t, 2048, 512, f1b1, nullptr, 0.f, nullptr, t16, 1);
  gemm(t16, f1w2t, 512, 2048, f1b2, x, 0.5f, xa, nullptr, 0);

  // ---- multi-head self-attention ----
  ln_kernel<<<M, 128, 0, stream>>>(xa, n2g, n2b, h16, nullptr);
  gemm(h16, wqt, 512, 512, bq, nullptr, 0.f, nullptr, q16, 0);
  gemm(h16, wkt, 512, 512, bk, nullptr, 0.f, nullptr, k16, 0);
  gemm(h16, wvt, 512, 512, bv, nullptr, 0.f, nullptr, v16, 0);
  flash_attn_kernel<<<dim3(64, 64), 32, 0, stream>>>(q16, k16, v16, o16);
  gemm(o16, wot, 512, 512, bo, xa, 1.0f, xa, nullptr, 0);

  // ---- convolution module ----
  ln_kernel<<<M, 128, 0, stream>>>(xa, n3g, n3b, h16, h32);
  ln_kernel<<<M, 128, 0, stream>>>(h32, clng, clnb, h16, nullptr);
  gemm(h16, pw1t, 1024, 512, pw1b, nullptr, 0.f, hg, nullptr, 0);
  glu_kernel<<<ew / 256, 256, 0, stream>>>(hg, ub);
  dwconv_bn_silu_kernel<<<ew / 256, 256, 0, stream>>>(ub, dww, dwb,
                                                      bng, bnbp, bnm, bnv, c16);
  gemm(c16, pw2t, 512, 512, pw2b, xa, 1.0f, xa, nullptr, 0);

  // ---- half-step FF2 ----
  ln_kernel<<<M, 128, 0, stream>>>(xa, n4g, n4b, h16, nullptr);
  gemm(h16, f2w1t, 2048, 512, f2b1, nullptr, 0.f, nullptr, t16, 1);
  gemm(t16, f2w2t, 512, 2048, f2b2, xa, 0.5f, outp, nullptr, 0);
}
